// model_cde_P_27144193310865
// MI455X (gfx1250) — matrix operand order repaired and checked by reading, not compiled, not run on hardware
//
#include <hip/hip_runtime.h>

typedef _Float16 half_t;
typedef __attribute__((ext_vector_type(16))) _Float16 v16h;
typedef __attribute__((ext_vector_type(8)))  float    v8f;

#define LSTEPS  64
#define CDIM    7
#define NSTEP   126      // (L-1)*K
#define HSTEP   0.5f
#define ZSTRIDE 34       // padded row stride for 16x32 f32 buffers (bank-conflict avoidance)
#define GSTRIDE 115      // padded row stride for 16x112 g buffer (odd -> conflict-free column reads)

#define LOG2E 1.44269504088896340736f
#define LN2   0.69314718055994530942f

__device__ __forceinline__ float softplus_f(float v) {
    // logaddexp(v,0) = max(v,0) + ln(1+exp(-|v|)); log arg in [1,2] -> raw v_log_f32 is safe
    float e = __builtin_amdgcn_exp2f(-fabsf(v) * LOG2E);      // exp(-|v|), arg <= 0 (no overflow)
    return fmaxf(v, 0.f) + LN2 * __builtin_amdgcn_logf(1.f + e);
}

__device__ __forceinline__ float tanh_f(float v) {
#if __has_builtin(__builtin_amdgcn_tanhf)
    return __builtin_amdgcn_tanhf(v);                          // native V_TANH_F32 (CDNA5 trans op)
#else
    float r;
    asm volatile("v_tanh_f32 %0, %1" : "=v"(r) : "v"(v));
    return r;
#endif
}

__launch_bounds__(64)
__global__ void cde_fused_kernel(const float* __restrict__ x,    // (B,64,7)
                                 const float* __restrict__ Wi,   // (32,7)
                                 const float* __restrict__ bi,   // (32)
                                 const float* __restrict__ Wf1,  // (32,32)
                                 const float* __restrict__ bf1,  // (32)
                                 const float* __restrict__ Wf2,  // (224,32)
                                 const float* __restrict__ bf2,  // (224)
                                 const float* __restrict__ Wo1,  // (16,32)
                                 const float* __restrict__ bo1,  // (16)
                                 const float* __restrict__ Wo2,  // (3,16)
                                 const float* __restrict__ bo2,  // (3)
                                 float* __restrict__ out)        // (B,3)
{
    __shared__ float diffs_s[63 * 16 * 7];     // [interval][row][c]
    __shared__ float zs    [16 * ZSTRIDE];     // current z (16 x 32)
    __shared__ float zstage[16 * ZSTRIDE];     // RK4 intermediate state
    __shared__ float h1s   [16 * ZSTRIDE];     // softplus(GEMM1) (16 x 32)
    __shared__ float dxs   [3 * 16 * 7];       // dXdt at {t, t+h/2, t+h}
    __shared__ float gs    [2 * 16 * GSTRIDE]; // per-wave tanh(GEMM2) (16 x 112)

    const int tid = threadIdx.x;      // 0..63 (2 wave32)
    const int w   = tid >> 5;         // wave id
    const int l   = tid & 31;         // lane id
    const int b0  = blockIdx.x * 16;  // batch tile base

    // ---- stage spline diffs into LDS: diffs[i][m][c] = x[m][i+1][c]-x[m][i][c] ----
    for (int e = tid; e < 63 * 112; e += 64) {
        int i = e / 112, r = e % 112, m = r / 7, c = r % 7;
        const float* xr = x + (size_t)(b0 + m) * (LSTEPS * CDIM);
        diffs_s[e] = xr[(i + 1) * CDIM + c] - xr[i * CDIM + c];
    }
    // ---- z0 = x[:,0,:] @ Wi^T + bi ----
    for (int e = tid; e < 16 * 32; e += 64) {
        int m = e >> 5, j = e & 31;
        const float* xr = x + (size_t)(b0 + m) * (LSTEPS * CDIM);
        float acc = bi[j];
        #pragma unroll
        for (int c = 0; c < 7; ++c) acc += xr[c] * Wi[j * 7 + c];
        zs[m * ZSTRIDE + j] = acc;
    }

    // ---- preload weights in WMMA B-operand layout (lane = column, K split by half-wave) ----
    const int ncol  = l & 15;
    const int khalf = (l < 16) ? 0 : 16;
    v16h bW1;
    v8f  cbias1;                       // bias as loop-invariant WMMA C operand
    {
        int j = 16 * w + ncol;
        #pragma unroll
        for (int i = 0; i < 16; ++i) bW1[i] = (half_t)Wf1[j * 32 + (khalf >> 1) + i + ((i >= 8) ? 8 : 0)];
        float bb1 = bf1[j];
        #pragma unroll
        for (int r = 0; r < 8; ++r) cbias1[r] = bb1;
    }
    v16h bW2[7];
    v8f  cbias2[7];
    #pragma unroll
    for (int u = 0; u < 7; ++u) {
        int mcol = 16 * (7 * w + u) + ncol;
        #pragma unroll
        for (int i = 0; i < 16; ++i) bW2[u][i] = (half_t)Wf2[mcol * 32 + (khalf >> 1) + i + ((i >= 8) ? 8 : 0)];
        float bb2 = bf2[mcol];
        #pragma unroll
        for (int r = 0; r < 8; ++r) cbias2[u][r] = bb2;
    }

    __syncthreads();

    // lane ownership: (row mrow) x (8 h-columns starting at hbase)
    const int mrow  = l & 15;
    const int hbase = 16 * w + 8 * (l >> 4);
    const int rbase = (l < 16) ? 0 : 8;    // D-tile row base
    const int arow  = l & 15;              // A-operand row
    const int ak    = (l < 16) ? 0 : 8;    // A-operand K chunk base

    float z_reg[8];
    #pragma unroll
    for (int j = 0; j < 8; ++j) z_reg[j] = zs[mrow * ZSTRIDE + hbase + j];

    #pragma unroll 1
    for (int step = 0; step < NSTEP; ++step) {
        const float t0 = (float)step * HSTEP;
        float kacc[8];
        #pragma unroll
        for (int j = 0; j < 8; ++j) kacc[j] = 0.f;

        for (int s = 0; s < 4; ++s) {
            const float* zsrc = (s == 0) ? zs : zstage;

            // ---- stage 1: A = f16(zsrc); GEMM1 tile w; softplus -> h1s ----
            v16h a;
            #pragma unroll
            for (int j = 0; j < 8; ++j) a[j]     = (half_t)zsrc[arow * ZSTRIDE + ak + j];
            #pragma unroll
            for (int j = 0; j < 8; ++j) a[8 + j] = (half_t)zsrc[arow * ZSTRIDE + ak + 16 + j];
            v8f d1 = __builtin_amdgcn_wmma_f32_16x16x32_f16(false, a, false, bW1,
                                                            (short)0, cbias1, false, false);
            #pragma unroll
            for (int r = 0; r < 8; ++r)
                h1s[(rbase + r) * ZSTRIDE + 16 * w + ncol] = softplus_f(d1[r]);

            if (s == 0) {  // dXdt at the 3 times of this RK4 step
                for (int e = tid; e < 3 * 112; e += 64) {
                    int si = e / 112, r = e % 112;
                    float tau = t0 + 0.25f * (float)si;
                    int idx = (int)tau; if (idx > 62) idx = 62;
                    float f  = tau - (float)idx;
                    int  i1  = (idx > 0) ? (idx - 1) : 0;
                    float D1 = diffs_s[i1 * 112 + r];
                    float D2 = diffs_s[idx * 112 + r];
                    dxs[e] = D1 + f * (4.f - 3.f * f) * (D2 - D1);
                }
            }
            __syncthreads();

            // ---- stage 2: A = f16(h1); 7x GEMM2 tiles; tanh -> gs; einsum -> k ----
            v16h ah;
            #pragma unroll
            for (int j = 0; j < 8; ++j) ah[j]     = (half_t)h1s[arow * ZSTRIDE + ak + j];
            #pragma unroll
            for (int j = 0; j < 8; ++j) ah[8 + j] = (half_t)h1s[arow * ZSTRIDE + ak + 16 + j];
            float* gw = gs + w * 16 * GSTRIDE;
            #pragma unroll
            for (int u = 0; u < 7; ++u) {
                v8f d2 = __builtin_amdgcn_wmma_f32_16x16x32_f16(false, ah, false, bW2[u],
                                                                (short)0, cbias2[u], false, false);
                #pragma unroll
                for (int r = 0; r < 8; ++r)
                    gw[(rbase + r) * GSTRIDE + 16 * u + ncol] = tanh_f(d2[r]);
            }
            // einsum('bhc,bc->bh'): wave-local through LDS (same-wave DS ops are in-order)
            const int dsel = (s == 0) ? 0 : ((s == 3) ? 2 : 1);
            float dxv[7];
            #pragma unroll
            for (int c = 0; c < 7; ++c) dxv[c] = dxs[dsel * 112 + mrow * 7 + c];
            float kv[8];
            #pragma unroll
            for (int j = 0; j < 8; ++j) {
                int hloc = 8 * (l >> 4) + j;
                const float* grow = gw + mrow * GSTRIDE + 7 * hloc;
                float acc = 0.f;
                #pragma unroll
                for (int c = 0; c < 7; ++c) acc += grow[c] * dxv[c];
                kv[j] = acc;
            }

            const float wgt = (s == 0 || s == 3) ? 1.f : 2.f;
            #pragma unroll
            for (int j = 0; j < 8; ++j) kacc[j] += wgt * kv[j];

            if (s < 3) {
                const float aco = (s == 2) ? HSTEP : (0.5f * HSTEP);
                #pragma unroll
                for (int j = 0; j < 8; ++j)
                    zstage[mrow * ZSTRIDE + hbase + j] = z_reg[j] + aco * kv[j];
            } else {
                #pragma unroll
                for (int j = 0; j < 8; ++j) {
                    z_reg[j] += (HSTEP / 6.f) * kacc[j];
                    zs[mrow * ZSTRIDE + hbase + j] = z_reg[j];
                }
            }
            __syncthreads();
        }
    }

    // ---- readout: softplus(zT @ Wo1^T + bo1) @ Wo2^T + bo2 ----
    if (tid < 16) {
        const int m = tid;
        float zrow[32];
        #pragma unroll
        for (int i = 0; i < 32; ++i) zrow[i] = zs[m * ZSTRIDE + i];
        float p[16];
        #pragma unroll
        for (int j = 0; j < 16; ++j) {
            float acc = bo1[j];
            #pragma unroll
            for (int i = 0; i < 32; ++i) acc += zrow[i] * Wo1[j * 32 + i];
            p[j] = softplus_f(acc);
        }
        #pragma unroll
        for (int o = 0; o < 3; ++o) {
            float acc = bo2[o];
            #pragma unroll
            for (int j = 0; j < 16; ++j) acc += p[j] * Wo2[o * 16 + j];
            out[(size_t)(b0 + m) * 3 + o] = acc;
        }
    }
}

extern "C" void kernel_launch(void* const* d_in, const int* in_sizes, int n_in,
                              void* d_out, int out_size, void* d_ws, size_t ws_size,
                              hipStream_t stream) {
    (void)in_sizes; (void)n_in; (void)d_ws; (void)ws_size; (void)out_size;
    const float* x   = (const float*)d_in[0];
    const float* Wi  = (const float*)d_in[1];
    const float* bi  = (const float*)d_in[2];
    const float* Wf1 = (const float*)d_in[3];
    const float* bf1 = (const float*)d_in[4];
    const float* Wf2 = (const float*)d_in[5];
    const float* bf2 = (const float*)d_in[6];
    const float* Wo1 = (const float*)d_in[7];
    const float* bo1 = (const float*)d_in[8];
    const float* Wo2 = (const float*)d_in[9];
    const float* bo2 = (const float*)d_in[10];
    float* out = (float*)d_out;
    // 4096 batch rows / 16 rows per block = 256 blocks, 2 wave32 per block
    cde_fused_kernel<<<256, 64, 0, stream>>>(x, Wi, bi, Wf1, bf1, Wf2, bf2,
                                             Wo1, bo1, Wo2, bo2, out);
}
